// AlignBinary_8718783610999
// MI455X (gfx1250) — 8-bit matrix operand order repaired and checked by reading, compiled, not run on hardware
//
#include <hip/hip_runtime.h>
#include <hip/hip_bf16.h>


typedef __attribute__((ext_vector_type(8))) int v8i;

#define B_DIM 128
#define L_DIM 128

__device__ __forceinline__ v8i make_a_frag(unsigned d0, unsigned d1, unsigned h) {
    v8i f;
#pragma unroll
    for (int v = 0; v < 8; ++v) {
        const unsigned g  = (unsigned)(v & 3);
        const unsigned kb = 16u * h + g * 4u;
        const unsigned tgt = (v >= 4) ? d1 : d0;
        const unsigned d   = tgt - kb;
        f[v] = (d < 4u) ? (int)(1u << (8u * (d & 3u))) : 0;
    }
    return f;
}

__device__ __forceinline__ v8i make_b_frag(unsigned d0, unsigned d1, unsigned h) {
    v8i f;
#pragma unroll
    for (int v = 0; v < 8; ++v) {
        const unsigned kb  = 16u * h + (unsigned)(v & 3) * 4u;
        const unsigned tgt = (v >= 4) ? d1 : d0;
        const unsigned d   = tgt - kb;
        f[v] = (d < 4u) ? (int)(1u << (8u * (d & 3u))) : 0;
    }
    return f;
}

__device__ __forceinline__ void split_digits(int id, unsigned& d0, unsigned& d1) {
    const bool valid = (id > 0);
    d0 = valid ? (unsigned)(id & 31) : 0x40000000u;
    d1 = valid ? (unsigned)(id >> 5) : 0x40000000u;
}

__global__ __launch_bounds__(256) void
align_binary_eq_wmma(const int* __restrict__ qry,
                     const int* __restrict__ cnd,
                     float* __restrict__ out) {
    const unsigned lane = threadIdx.x & 31u;
    const unsigned wave = threadIdx.x >> 5;
    const unsigned h    = lane >> 4;
    const unsigned m    = lane & 15u;

    const unsigned wg  = blockIdx.x * 8u + wave;
    const unsigned t0  = wg * 2u;
    const unsigned b   = t0 >> 6;
    const unsigned rt  = (t0 >> 3) & 7u;
    const unsigned ct0 = t0 & 7u;

    const int idq = qry[b * L_DIM + rt * 16u + m];
    unsigned dq0, dq1;
    split_digits(idq, dq0, dq1);
    const v8i a = make_a_frag(dq0, dq1, h);

#pragma unroll
    for (unsigned t = 0; t < 2; ++t) {
        const unsigned ct = ct0 + t;

        const int idc = cnd[b * L_DIM + ct * 16u + m];
        unsigned dc0, dc1;
        split_digits(idc, dc0, dc1);
        const v8i bf = make_b_frag(dc0, dc1, h);

        v8i c = {0, 0, 0, 0, 0, 0, 0, 0};
        c = __builtin_amdgcn_wmma_i32_16x16x64_iu8(false, a, false, bf, c,
                                                   false, false);

#pragma unroll
        for (int r = 0; r < 8; ++r) {
            const unsigned mm  = (unsigned)r + 8u * h;
            const float    val = (c[r] == 2) ? 1.0f : 0.0f;
            out[(b * L_DIM + rt * 16u + mm) * L_DIM + ct * 16u + m] = val;
        }
    }
}

extern "C" void kernel_launch(void* const* d_in, const int* in_sizes, int n_in,
                              void* d_out, int out_size, void* d_ws, size_t ws_size,
                              hipStream_t stream) {
    (void)in_sizes; (void)n_in; (void)d_ws; (void)ws_size; (void)out_size;
    const int* qry = (const int*)d_in[1];
    const int* cnd = (const int*)d_in[2];
    float*     out = (float*)d_out;

    dim3 grid(512), block(256);
    align_binary_eq_wmma<<<grid, block, 0, stream>>>(qry, cnd, out);
}
